// TransformerEncoder_22746146799764
// MI455X (gfx1250) — matrix operand order repaired and checked by reading, not compiled, not run on hardware
//
#include <hip/hip_runtime.h>
#include <hip/hip_bf16.h>
#include <stdint.h>

// ---------------- model dims ----------------
#define LAY 6
#define DMODEL 1024
#define HEADS 16
#define DHEAD 64
#define FFN 4096
#define SEQ 1024
#define BATCH 2
#define TOK (BATCH * SEQ)   // 2048

typedef __bf16 bf16_t;
typedef __attribute__((ext_vector_type(8)))  __bf16 bf16x8;
typedef __attribute__((ext_vector_type(16))) __bf16 bf16x16;
typedef __attribute__((ext_vector_type(8)))  float  f32x8;

__device__ __forceinline__ float gelu_f(float x) {
    float x3 = x * x * x;
    return 0.5f * x * (1.0f + tanhf(0.7978845608028654f * (x + 0.044715f * x3)));
}

// gfx1250 async copy: global memory -> LDS, 16B per lane, tracked by ASYNCcnt.
__device__ __forceinline__ void async_ld16(uint32_t lds_off, const void* gaddr) {
    asm volatile("global_load_async_to_lds_b128 %0, %1, off"
                 :: "v"(lds_off), "v"(gaddr) : "memory");
}
__device__ __forceinline__ void wait_async0() {
    asm volatile("s_wait_asynccnt 0" ::: "memory");
}

// ---------------------------------------------------------------------------
// LDS-staged, double-buffered bf16 WMMA GEMM:
//   C[m,n] = alpha * sum_k A[m,k] * W[n,k]  (+bias) (opt GELU) (opt +residual)
// Block = WAVES_M*WAVES_N waves (always 128 threads in our configs).
// Block tile BM x BN = (32*WAVES_M) x (64*WAVES_N); wave tile 32x64
// (2x4 fragments -> 8 v_wmma_f32_16x16x32_bf16 per 32-wide K step).
// Tiles are filled with global_load_async_to_lds_b128 (ASYNCcnt) and
// double-buffered in LDS. All dims must divide the tiles exactly.
// ---------------------------------------------------------------------------
template<int WAVES_M, int WAVES_N>
__global__ __launch_bounds__(32 * WAVES_M * WAVES_N) void gemm_wmma_bf16(
    const bf16_t* __restrict__ A, long long aZ, int lda,
    const bf16_t* __restrict__ W, long long wZ, int ldw,
    const float* __restrict__ bias,
    const float* residual,
    float* outF, __hip_bfloat16* outB,
    int ldc, long long cZ, int K, float alpha, int doGelu)
{
    constexpr int BM = 32 * WAVES_M;
    constexpr int BN = 64 * WAVES_N;
    constexpr int NT = 32 * WAVES_M * WAVES_N;   // threads (128)
    constexpr int AE = BM * 32;                  // tile elements
    constexpr int BE = BN * 32;

    __shared__ alignas(64) bf16_t tileA[2 * AE];
    __shared__ alignas(64) bf16_t tileB[2 * BE];

    const int z = blockIdx.z;
    A += (size_t)z * (size_t)aZ;
    W += (size_t)z * (size_t)wZ;
    const size_t cOff = (size_t)z * (size_t)cZ;

    const int tid  = threadIdx.x;
    const int lane = tid & 31;
    const int wv   = tid >> 5;
    const int wm   = wv / WAVES_N;
    const int wn   = wv % WAVES_N;
    const int mBlk = blockIdx.y * BM;
    const int nBlk = blockIdx.x * BN;
    const int m0w  = wm * 32;                    // wave tile origin in block tile
    const int n0w  = wn * 64;
    const int lr   = lane & 15;
    const int lh   = lane >> 4;

    const uint32_t ldsA0 = (uint32_t)(uintptr_t)&tileA[0];
    const uint32_t ldsB0 = (uint32_t)(uintptr_t)&tileB[0];

    // issue async copy of one 32-deep K slab into LDS buffer `buf`
    auto issue = [&](int kk, int buf) {
        const uint32_t aB = ldsA0 + (uint32_t)buf * (AE * 2);
        const uint32_t bB = ldsB0 + (uint32_t)buf * (BE * 2);
        #pragma unroll
        for (int c = tid; c < BM * 4; c += NT) {       // A: BM rows x 64B
            const int row = c >> 2, cc = c & 3;
            async_ld16(aB + (uint32_t)(row * 64 + cc * 16),
                       A + (size_t)(mBlk + row) * lda + kk + cc * 8);
        }
        #pragma unroll
        for (int c = tid; c < BN * 4; c += NT) {       // B: BN rows x 64B
            const int row = c >> 2, cc = c & 3;
            async_ld16(bB + (uint32_t)(row * 64 + cc * 16),
                       W + (size_t)(nBlk + row) * ldw + kk + cc * 8);
        }
    };

    f32x8 acc[2][4] = {};

    issue(0, 0);
    const int nTiles = K >> 5;
    for (int kt = 0; kt < nTiles; ++kt) {
        const int cur = kt & 1;
        wait_async0();        // this wave's async copies into `cur` complete
        __syncthreads();      // all waves' copies complete / prev reads done
        if (kt + 1 < nTiles) issue((kt + 1) << 5, cur ^ 1);

        const bf16_t* tA = tileA + cur * AE;
        const bf16_t* tB = tileB + cur * BE;

        bf16x16 afrag[2], bfrag[4];
        #pragma unroll
        for (int i = 0; i < 2; ++i) {
            // A 16x32 fragment, contiguous k order (same as B): lanes 0-15 K[0..15]; lanes 16-31 K[16..31]
            afrag[i] = *(const bf16x16*)(tA + (m0w + 16 * i + lr) * 32 + 16 * lh);
        }
        #pragma unroll
        for (int j = 0; j < 4; ++j) {
            // B[k,n] = W[n,k]: lane n<16 -> col n K[0..15]; lane n+16 -> K[16..31]
            bfrag[j] = *(const bf16x16*)(tB + (n0w + 16 * j + lr) * 32 + 16 * lh);
        }
        #pragma unroll
        for (int i = 0; i < 2; ++i)
            #pragma unroll
            for (int j = 0; j < 4; ++j)
                acc[i][j] = __builtin_amdgcn_wmma_f32_16x16x32_bf16(
                    false, afrag[i], false, bfrag[j],
                    (short)0, acc[i][j], false, false);
        __syncthreads();
    }

    // C/D layout: VGPR r -> M = r (+8 for lanes 16-31), N = lane&15
    #pragma unroll
    for (int i = 0; i < 2; ++i) {
        #pragma unroll
        for (int j = 0; j < 4; ++j) {
            const int n = nBlk + n0w + 16 * j + lr;
            #pragma unroll
            for (int r = 0; r < 8; ++r) {
                const int m = mBlk + m0w + 16 * i + 8 * lh + r;
                float v = acc[i][j][r] * alpha;
                if (bias)   v += bias[n];
                if (doGelu) v = gelu_f(v);
                const size_t ci = cOff + (size_t)m * ldc + n;
                if (residual) v += residual[ci];
                if (outF) outF[ci] = v;
                if (outB) outB[ci] = __float2bfloat16(v);
            }
        }
    }
}

// ---------------------------------------------------------------------------
// x[t,d] = emb[ids[t],d] * sqrt(D) + pos[t%S, d]
// ---------------------------------------------------------------------------
__global__ __launch_bounds__(256) void embed_kernel(
    const int* __restrict__ ids, const float* __restrict__ emb,
    const float* __restrict__ pos, float* __restrict__ x)
{
    const size_t idx = (size_t)blockIdx.x * 256 + threadIdx.x;
    const int t = (int)(idx / DMODEL);
    const int d = (int)(idx % DMODEL);
    const int s = t % SEQ;
    x[idx] = emb[(size_t)ids[t] * DMODEL + d] * 32.0f + pos[(size_t)s * DMODEL + d];
}

__global__ __launch_bounds__(256) void cast_f32_bf16(
    const float* __restrict__ src, __hip_bfloat16* __restrict__ dst)
{
    const size_t idx = (size_t)blockIdx.x * 256 + threadIdx.x;
    dst[idx] = __float2bfloat16(src[idx]);
}

// ---------------------------------------------------------------------------
// LayerNorm over D=1024, one block (256 threads) per token, bf16 output.
// ---------------------------------------------------------------------------
__global__ __launch_bounds__(256) void ln_kernel(
    const float* __restrict__ x, const float* __restrict__ w,
    const float* __restrict__ b, __hip_bfloat16* __restrict__ out)
{
    __shared__ float red[256];
    const int t = blockIdx.x, tid = threadIdx.x;
    const float4 v = ((const float4*)(x + (size_t)t * DMODEL))[tid];

    red[tid] = v.x + v.y + v.z + v.w;
    __syncthreads();
    for (int s = 128; s > 0; s >>= 1) { if (tid < s) red[tid] += red[tid + s]; __syncthreads(); }
    const float mean = red[0] * (1.0f / DMODEL);
    __syncthreads();

    const float d0 = v.x - mean, d1 = v.y - mean, d2 = v.z - mean, d3 = v.w - mean;
    red[tid] = d0 * d0 + d1 * d1 + d2 * d2 + d3 * d3;
    __syncthreads();
    for (int s = 128; s > 0; s >>= 1) { if (tid < s) red[tid] += red[tid + s]; __syncthreads(); }
    const float rstd = rsqrtf(red[0] * (1.0f / DMODEL) + 1e-5f);

    const float4 wv = ((const float4*)w)[tid];
    const float4 bv = ((const float4*)b)[tid];
    __hip_bfloat16* o = out + (size_t)t * DMODEL + 4 * tid;
    o[0] = __float2bfloat16(wv.x * d0 * rstd + bv.x);
    o[1] = __float2bfloat16(wv.y * d1 * rstd + bv.y);
    o[2] = __float2bfloat16(wv.z * d2 * rstd + bv.z);
    o[3] = __float2bfloat16(wv.w * d3 * rstd + bv.w);
}

// ---------------------------------------------------------------------------
// q,k fp32 [T,D] -> bf16 [b,h,s,dh]; v fp32 -> vt bf16 [b,h,dh,s]
// ---------------------------------------------------------------------------
__global__ __launch_bounds__(256) void rearrange_qkv(
    const float* __restrict__ q, const float* __restrict__ k,
    const float* __restrict__ v,
    __hip_bfloat16* __restrict__ qh, __hip_bfloat16* __restrict__ kh,
    __hip_bfloat16* __restrict__ vt)
{
    const size_t idx = (size_t)blockIdx.x * 256 + threadIdx.x;
    const int t = (int)(idx / DMODEL);
    const int c = (int)(idx % DMODEL);
    const int b = t / SEQ, s = t % SEQ;
    const int h = c / DHEAD, dh = c % DHEAD;
    const size_t bh = (size_t)b * HEADS + h;
    qh[(bh * SEQ + s) * DHEAD + dh] = __float2bfloat16(q[idx]);
    kh[(bh * SEQ + s) * DHEAD + dh] = __float2bfloat16(k[idx]);
    vt[(bh * DHEAD + dh) * SEQ + s] = __float2bfloat16(v[idx]);
}

// ---------------------------------------------------------------------------
// Softmax over a row of S=1024 fp32 scores; bf16 result written IN PLACE over
// the same row storage. One block (256 threads) per row. mask: 1 byte per key.
// ---------------------------------------------------------------------------
__global__ __launch_bounds__(256) void softmax_kernel(
    float* __restrict__ scores, const unsigned char* __restrict__ mask)
{
    __shared__ float red[256];
    const size_t row = blockIdx.x;
    float* sr = scores + row * SEQ;
    const int tid = threadIdx.x;

    const float4 v = ((const float4*)sr)[tid];
    const unsigned char* mp = mask + 4 * tid;
    const float ninf = -__builtin_inff();
    float x0 = mp[0] ? v.x : ninf;
    float x1 = mp[1] ? v.y : ninf;
    float x2 = mp[2] ? v.z : ninf;
    float x3 = mp[3] ? v.w : ninf;

    red[tid] = fmaxf(fmaxf(x0, x1), fmaxf(x2, x3));
    __syncthreads();
    for (int s = 128; s > 0; s >>= 1) { if (tid < s) red[tid] = fmaxf(red[tid], red[tid + s]); __syncthreads(); }
    const float mx = red[0];
    __syncthreads();

    const float e0 = __expf(x0 - mx), e1 = __expf(x1 - mx);
    const float e2 = __expf(x2 - mx), e3 = __expf(x3 - mx);
    red[tid] = e0 + e1 + e2 + e3;
    __syncthreads();
    for (int s = 128; s > 0; s >>= 1) { if (tid < s) red[tid] += red[tid + s]; __syncthreads(); }
    const float inv = 1.0f / red[0];
    __syncthreads();   // all row reads complete before overwriting as bf16

    __hip_bfloat16* ob = (__hip_bfloat16*)sr;
    ob[4 * tid + 0] = __float2bfloat16(e0 * inv);
    ob[4 * tid + 1] = __float2bfloat16(e1 * inv);
    ob[4 * tid + 2] = __float2bfloat16(e2 * inv);
    ob[4 * tid + 3] = __float2bfloat16(e3 * inv);
}

// ---------------------------------------------------------------------------
// workspace layout (bytes)
// ---------------------------------------------------------------------------
#define MB (1ull << 20)
#define OFF_X   (0 * MB)     // fp32 [T,D]            8 MB
#define OFF_Q   (8 * MB)     // fp32 [T,D] q / attn-o 8 MB
#define OFF_K   (16 * MB)    // fp32 [T,D]            8 MB
#define OFF_V   (24 * MB)    // fp32 [T,D]            8 MB
#define OFF_HBF (32 * MB)    // bf16 [T,D] h/obf/h2   4 MB
#define OFF_QH  (36 * MB)    // bf16 [B,H,S,DH]       4 MB
#define OFF_KH  (40 * MB)    // bf16 [B,H,S,DH]       4 MB
#define OFF_VT  (44 * MB)    // bf16 [B,H,DH,S]       4 MB
#define OFF_WQ  (48 * MB)    // bf16 [D,D]            2 MB
#define OFF_WK  (50 * MB)
#define OFF_WV  (52 * MB)
#define OFF_WO  (54 * MB)
#define OFF_W1  (56 * MB)    // bf16 [F,D]            8 MB
#define OFF_W2  (64 * MB)    // bf16 [D,F]            8 MB
#define OFF_Y1  (72 * MB)    // bf16 [T,F]           16 MB
#define OFF_SC  (88 * MB)    // fp32 [H,S,S] chunk   64 MB  (total 152 MB)

extern "C" void kernel_launch(void* const* d_in, const int* in_sizes, int n_in,
                              void* d_out, int out_size, void* d_ws, size_t ws_size,
                              hipStream_t stream)
{
    (void)in_sizes; (void)n_in; (void)out_size; (void)ws_size;

    const int*           ids  = (const int*)d_in[0];
    const unsigned char* mask = (const unsigned char*)d_in[1];
    const float* emb  = (const float*)d_in[2];
    const float* pos  = (const float*)d_in[3];
    const float* ln1w = (const float*)d_in[4];
    const float* ln1b = (const float*)d_in[5];
    const float* qw   = (const float*)d_in[6];
    const float* qb   = (const float*)d_in[7];
    const float* kw   = (const float*)d_in[8];
    const float* kb   = (const float*)d_in[9];
    const float* vw   = (const float*)d_in[10];
    const float* vb   = (const float*)d_in[11];
    const float* ow   = (const float*)d_in[12];
    const float* ob   = (const float*)d_in[13];
    const float* ln2w = (const float*)d_in[14];
    const float* ln2b = (const float*)d_in[15];
    const float* w1   = (const float*)d_in[16];
    const float* b1   = (const float*)d_in[17];
    const float* w2   = (const float*)d_in[18];
    const float* b2   = (const float*)d_in[19];

    char* wsb = (char*)d_ws;
    float*          x    = (float*)(wsb + OFF_X);
    float*          qf   = (float*)(wsb + OFF_Q);   // also attention output o
    float*          kf   = (float*)(wsb + OFF_K);
    float*          vf   = (float*)(wsb + OFF_V);
    __hip_bfloat16* hbf  = (__hip_bfloat16*)(wsb + OFF_HBF);
    __hip_bfloat16* qh   = (__hip_bfloat16*)(wsb + OFF_QH);
    __hip_bfloat16* kh   = (__hip_bfloat16*)(wsb + OFF_KH);
    __hip_bfloat16* vt   = (__hip_bfloat16*)(wsb + OFF_VT);
    bf16_t*         wqB  = (bf16_t*)(wsb + OFF_WQ);
    bf16_t*         wkB  = (bf16_t*)(wsb + OFF_WK);
    bf16_t*         wvB  = (bf16_t*)(wsb + OFF_WV);
    bf16_t*         woB  = (bf16_t*)(wsb + OFF_WO);
    bf16_t*         w1B  = (bf16_t*)(wsb + OFF_W1);
    bf16_t*         w2B  = (bf16_t*)(wsb + OFF_W2);
    bf16_t*         y1   = (bf16_t*)(wsb + OFF_Y1);
    float*          sc   = (float*)(wsb + OFF_SC);

    const int eltBlocks = (TOK * DMODEL) / 256;   // 8192

    embed_kernel<<<eltBlocks, 256, 0, stream>>>(ids, emb, pos, x);

    // big tiles: block 64x128 (waves 2x2)
    auto gemm = [&](const void* A, long long aZ, int lda,
                    const void* W, long long wZ, int ldw,
                    const float* bias, const float* resid,
                    float* outF, __hip_bfloat16* outB,
                    int ldc, long long cZ, int M, int N, int K, int Z,
                    float alpha, int doGelu) {
        dim3 g(N / 128, M / 64, Z);
        gemm_wmma_bf16<2, 2><<<g, 128, 0, stream>>>(
            (const bf16_t*)A, aZ, lda, (const bf16_t*)W, wZ, ldw,
            bias, resid, outF, outB, ldc, cZ, K, alpha, doGelu);
    };
    // narrow tiles: block 128x64 (waves 4x1) for N=64
    auto gemmN64 = [&](const void* A, long long aZ, int lda,
                       const void* W, long long wZ, int ldw,
                       float* outF, int ldc, long long cZ, int M, int K, int Z) {
        dim3 g(1, M / 128, Z);
        gemm_wmma_bf16<4, 1><<<g, 128, 0, stream>>>(
            (const bf16_t*)A, aZ, lda, (const bf16_t*)W, wZ, ldw,
            nullptr, nullptr, outF, nullptr, ldc, cZ, K, 1.0f, 0);
    };

    for (int l = 0; l < LAY; ++l) {
        const size_t lDD = (size_t)l * DMODEL * DMODEL;
        const size_t lFD = (size_t)l * FFN * DMODEL;
        // weight fp32 -> bf16 for this layer
        cast_f32_bf16<<<(DMODEL * DMODEL) / 256, 256, 0, stream>>>(qw + lDD, (__hip_bfloat16*)wqB);
        cast_f32_bf16<<<(DMODEL * DMODEL) / 256, 256, 0, stream>>>(kw + lDD, (__hip_bfloat16*)wkB);
        cast_f32_bf16<<<(DMODEL * DMODEL) / 256, 256, 0, stream>>>(vw + lDD, (__hip_bfloat16*)wvB);
        cast_f32_bf16<<<(DMODEL * DMODEL) / 256, 256, 0, stream>>>(ow + lDD, (__hip_bfloat16*)woB);
        cast_f32_bf16<<<(FFN * DMODEL) / 256, 256, 0, stream>>>(w1 + lFD, (__hip_bfloat16*)w1B);
        cast_f32_bf16<<<(FFN * DMODEL) / 256, 256, 0, stream>>>(w2 + lFD, (__hip_bfloat16*)w2B);

        // LN1 -> bf16 activations
        ln_kernel<<<TOK, 256, 0, stream>>>(x, ln1w + (size_t)l * DMODEL,
                                           ln1b + (size_t)l * DMODEL, hbf);

        // Q/K/V projections (fp32 out)
        gemm(hbf, 0, DMODEL, wqB, 0, DMODEL, qb + (size_t)l * DMODEL, nullptr,
             qf, nullptr, DMODEL, 0, TOK, DMODEL, DMODEL, 1, 1.0f, 0);
        gemm(hbf, 0, DMODEL, wkB, 0, DMODEL, kb + (size_t)l * DMODEL, nullptr,
             kf, nullptr, DMODEL, 0, TOK, DMODEL, DMODEL, 1, 1.0f, 0);
        gemm(hbf, 0, DMODEL, wvB, 0, DMODEL, vb + (size_t)l * DMODEL, nullptr,
             vf, nullptr, DMODEL, 0, TOK, DMODEL, DMODEL, 1, 1.0f, 0);

        rearrange_qkv<<<eltBlocks, 256, 0, stream>>>(qf, kf, vf, qh, kh, vt);

        // attention, one batch chunk at a time (scores buffer = [H,S,S])
        for (int b = 0; b < BATCH; ++b) {
            const size_t bhOff = (size_t)b * HEADS * SEQ * DHEAD;   // elements
            // scores = (Q K^T) / sqrt(DH)   [per head]
            gemm(qh + bhOff, (long long)SEQ * DHEAD, DHEAD,
                 kh + bhOff, (long long)SEQ * DHEAD, DHEAD,
                 nullptr, nullptr, sc, nullptr,
                 SEQ, (long long)SEQ * SEQ, SEQ, SEQ, DHEAD, HEADS, 0.125f, 0);
            // softmax rows, bf16 in place
            softmax_kernel<<<HEADS * SEQ, 256, 0, stream>>>(sc, mask + (size_t)b * SEQ);
            // o = attn @ V  -> fp32 written into [T,D] at column h*DH
            gemmN64((const bf16_t*)sc, 2ll * SEQ * SEQ, 2 * SEQ,
                    vt + (size_t)b * HEADS * DHEAD * SEQ, (long long)DHEAD * SEQ, SEQ,
                    qf + (size_t)b * SEQ * DMODEL, DMODEL, (long long)DHEAD,
                    SEQ, SEQ, HEADS);
        }

        // o -> bf16, O projection with residual into x
        cast_f32_bf16<<<eltBlocks, 256, 0, stream>>>(qf, hbf);
        gemm(hbf, 0, DMODEL, woB, 0, DMODEL, ob + (size_t)l * DMODEL, x,
             x, nullptr, DMODEL, 0, TOK, DMODEL, DMODEL, 1, 1.0f, 0);

        // MLP
        ln_kernel<<<TOK, 256, 0, stream>>>(x, ln2w + (size_t)l * DMODEL,
                                           ln2b + (size_t)l * DMODEL, hbf);
        gemm(hbf, 0, DMODEL, w1B, 0, DMODEL, b1 + (size_t)l * FFN, nullptr,
             nullptr, (__hip_bfloat16*)y1, FFN, 0, TOK, FFN, DMODEL, 1, 1.0f, 1);
        gemm(y1, 0, FFN, w2B, 0, FFN, b2 + (size_t)l * DMODEL, x,
             x, nullptr, DMODEL, 0, TOK, DMODEL, FFN, 1, 1.0f, 0);
    }

    hipMemcpyAsync(d_out, x, (size_t)TOK * DMODEL * sizeof(float),
                   hipMemcpyDeviceToDevice, stream);
}
